// Devign2_40458591929263
// MI455X (gfx1250) — matrix operand order repaired and checked by reading, not compiled, not run on hardware
//
#include <hip/hip_runtime.h>
#include <math.h>

// ---------------------------------------------------------------------------
// Devign GNN pipeline for MI455X (gfx1250, wave32, WMMA).
// GEMM operands pre-packed into padded f16 buffers; the WMMA GEMM uses a
// double-buffered LDS pipeline staged with GLOBAL_LOAD_ASYNC_TO_LDS_B128
// (ASYNCcnt) when the toolchain exposes the builtin, else b128 load/store.
// Workspace requirement: ~0.96 GB (see offsets in kernel_launch).
// ---------------------------------------------------------------------------

#define NODES   104960      // 512*205  (= 128 * 820)
#define NGRAPH  512
#define NPG     205
#define EDGES   839680
#define EMB     768
#define COMP    101
#define OUTC    200
#define KPAD    224         // OUTC padded to multiple of 32
#define GATES   600

typedef __attribute__((ext_vector_type(16))) _Float16 v16h;
typedef __attribute__((ext_vector_type(8)))  _Float16 v8h;
typedef __attribute__((ext_vector_type(8)))  float    v8f;
typedef __attribute__((__vector_size__(16))) int      i128;   // int4 for async-LDS builtin

#define AS1 __attribute__((address_space(1)))
#define AS3 __attribute__((address_space(3)))

#if defined(__has_builtin)
#if __has_builtin(__builtin_amdgcn_global_load_async_to_lds_b128) && \
    __has_builtin(__builtin_amdgcn_s_wait_asynccnt)
#define USE_ASYNC_LDS 1
#endif
#endif

// ---------------------------------------------------------------------------
// Utility kernels
// ---------------------------------------------------------------------------
__global__ __launch_bounds__(256)
void fill_kernel(float* __restrict__ p, float v, long long n)
{
    long long gid = (long long)blockIdx.x * blockDim.x + threadIdx.x;
    if (gid < n) p[gid] = v;
}

// fp32 -> f16, 8 elements per thread (n must be a multiple of 8)
__global__ __launch_bounds__(256)
void cvt_f16_kernel(const float* __restrict__ src, _Float16* __restrict__ dst,
                    long long n8)
{
    long long gid = (long long)blockIdx.x * blockDim.x + threadIdx.x;
    if (gid >= n8) return;
    const float4* s = (const float4*)src + gid * 2;
    float4 a = s[0], b = s[1];
    v8h o;
    o[0] = (_Float16)a.x; o[1] = (_Float16)a.y; o[2] = (_Float16)a.z; o[3] = (_Float16)a.w;
    o[4] = (_Float16)b.x; o[5] = (_Float16)b.y; o[6] = (_Float16)b.z; o[7] = (_Float16)b.w;
    ((v8h*)dst)[gid] = o;
}

// [NODES,200] fp32 -> [NODES,224] f16 (zero-padded K), 8 halves per thread
__global__ __launch_bounds__(256)
void cvt_pad224_kernel(const float* __restrict__ src, _Float16* __restrict__ dst)
{
    const long long total = (long long)NODES * (KPAD / 8);  // 28 chunks/row
    long long gid = (long long)blockIdx.x * blockDim.x + threadIdx.x;
    if (gid >= total) return;
    long long r = gid / (KPAD / 8);
    int c8 = (int)(gid - r * (KPAD / 8));
    v8h o = {};
    if (c8 < OUTC / 8) {                     // 200/8 = 25 full chunks
        const float4* s = (const float4*)(src + r * OUTC + c8 * 8);
        float4 a = s[0], b = s[1];
        o[0] = (_Float16)a.x; o[1] = (_Float16)a.y; o[2] = (_Float16)a.z; o[3] = (_Float16)a.w;
        o[4] = (_Float16)b.x; o[5] = (_Float16)b.y; o[6] = (_Float16)b.z; o[7] = (_Float16)b.w;
    }
    ((v8h*)(dst + r * KPAD))[c8] = o;
}

// Pack B^T: dst[n][k] = src[k*ldK + n*ldN] (f32->f16), zero-padded to Npad x Kpad
__global__ __launch_bounds__(256)
void pack_bt_kernel(const float* __restrict__ src, int ldK, int ldN,
                    _Float16* __restrict__ dst, int Kpad, int Npad, int K, int N)
{
    long long total = (long long)Npad * Kpad;
    long long gid = (long long)blockIdx.x * blockDim.x + threadIdx.x;
    if (gid >= total) return;
    int k = (int)(gid % Kpad);
    int n = (int)(gid / Kpad);
    float v = (n < N && k < K) ? src[(size_t)k * ldK + (size_t)n * ldN] : 0.0f;
    dst[gid] = (_Float16)v;
}

// ---------------------------------------------------------------------------
// WMMA GEMM: C[M x Ncols] = act(A * B + bias)
//   A : f16 [M x lda], M % 128 == 0, K % 32 == 0, unguarded rows
//   BT: f16 [Npad x ldb] (N-major / transposed B), Npad % 64 == 0
// 256 threads = 8 waves; 128x64 tile; each wave: 1 A-frag, 4 B-frags, 4 WMMAs.
// Double-buffered LDS (24 KB); async global->LDS staging when available.
// ---------------------------------------------------------------------------
#define GBM 128
#define GBN 64
#define GBK 32

__global__ __launch_bounds__(256)
void wmma_gemm_f16(const _Float16* __restrict__ A, int lda,
                   const _Float16* __restrict__ BT, int ldb,
                   const float* __restrict__ bias,
                   float* __restrict__ C, int ldc,
                   int K, int Ncols, int relu)
{
    __shared__ v8h As4[2][GBM][4];   // 2 x 128 rows x 32 halves (16 KB)
    __shared__ v8h Bs4[2][GBN][4];   // 2 x 64 N-rows x 32 halves (8 KB)

    const int tid  = threadIdx.x;
    const int lane = tid & 31;
    const int wid  = tid >> 5;            // 0..7 -> 16-row strip

    const int tileM = blockIdx.x * GBM;
    const int tileN = blockIdx.y * GBN;

    // Per-thread staging coordinates: 16B chunk (r, c8)
    const int r  = tid >> 2;              // 0..63
    const int c8 = tid & 3;               // 0..3
    const _Float16* gA0 = A  + (size_t)(tileM + r)      * lda + c8 * 8;
    const _Float16* gA1 = A  + (size_t)(tileM + r + 64) * lda + c8 * 8;
    const _Float16* gB0 = BT + (size_t)(tileN + r)      * ldb + c8 * 8;

    v8f acc[4] = {};

#if USE_ASYNC_LDS
    auto stage = [&](int k0, int buf) {
        __builtin_amdgcn_global_load_async_to_lds_b128(
            (AS1 i128*)(gA0 + k0), (AS3 i128*)&As4[buf][r][c8], 0, 0);
        __builtin_amdgcn_global_load_async_to_lds_b128(
            (AS1 i128*)(gA1 + k0), (AS3 i128*)&As4[buf][r + 64][c8], 0, 0);
        __builtin_amdgcn_global_load_async_to_lds_b128(
            (AS1 i128*)(gB0 + k0), (AS3 i128*)&Bs4[buf][r][c8], 0, 0);
    };
#else
    auto stage = [&](int k0, int buf) {
        v8h a0 = *(const v8h*)(gA0 + k0);    // issue all three loads first so
        v8h a1 = *(const v8h*)(gA1 + k0);    // the compiler can step loadcnt
        v8h b0 = *(const v8h*)(gB0 + k0);
        As4[buf][r][c8]      = a0;
        As4[buf][r + 64][c8] = a1;
        Bs4[buf][r][c8]      = b0;
    };
#endif

    auto compute = [&](int buf) {
        // A fragment: 16-bit A 16x32 layout (lanes 0-15: K 0-7/16-23,
        // lanes 16-31: K 8-15/24-31); chunks of 8 halves line up exactly.
        const int m  = wid * 16 + (lane & 15);
        const int lo = (lane < 16) ? 0 : 1;
        const int hi = (lane < 16) ? 2 : 3;
        v8h al = As4[buf][m][lo], ah = As4[buf][m][hi];
        v16h af = __builtin_shufflevector(al, ah,
                      0,1,2,3,4,5,6,7,8,9,10,11,12,13,14,15);
        // B fragments: N = lane%16, lanes 0-15 hold K 0-7/16-23, lanes 16-31 K 8-15/24-31
        const int ko = (lane < 16) ? 0 : 1;
        #pragma unroll
        for (int s = 0; s < 4; ++s) {
            const int n0 = s * 16 + (lane & 15);
            v8h bl = Bs4[buf][n0][ko], bh = Bs4[buf][n0][ko + 2];
            v16h bf = __builtin_shufflevector(bl, bh,
                          0,1,2,3,4,5,6,7,8,9,10,11,12,13,14,15);
            acc[s] = __builtin_amdgcn_wmma_f32_16x16x32_f16(false, af, false, bf,
                                                            (short)0, acc[s], false, false);
        }
    };

    int buf = 0;
    stage(0, 0);
#if USE_ASYNC_LDS
    for (int k0 = 0; k0 < K; k0 += GBK) {
        __builtin_amdgcn_s_wait_asynccnt(0);   // this wave's stage(buf) landed
        __syncthreads();                       // all waves' stage(buf) landed
        if (k0 + GBK < K) stage(k0 + GBK, buf ^ 1);  // overlaps with WMMAs
        compute(buf);
        buf ^= 1;
    }
#else
    for (int k0 = 0; k0 < K; k0 += GBK) {
        __syncthreads();                       // stage(buf) visible to all
        compute(buf);
        if (k0 + GBK < K) stage(k0 + GBK, buf ^ 1);
        buf ^= 1;
    }
#endif

    // C/D layout: VGPR r -> row r (lanes 0-15) / r+8 (lanes 16-31), col = lane%16
    const int rbase = tileM + wid * 16 + ((lane < 16) ? 0 : 8);
    const int cb    = lane & 15;
    #pragma unroll
    for (int s = 0; s < 4; ++s) {
        int gc = tileN + s * 16 + cb;
        if (gc >= Ncols) continue;
        float bv = bias ? bias[gc] : 0.0f;
        #pragma unroll
        for (int rr = 0; rr < 8; ++rr) {
            float v = acc[s][rr] + bv;
            if (relu) v = fmaxf(v, 0.0f);
            C[(size_t)(rbase + rr) * ldc + gc] = v;
        }
    }
}

// ---------------------------------------------------------------------------
// agg[dst] += m[src] over all edges; one float4 (of 50) per thread.
// ---------------------------------------------------------------------------
__global__ __launch_bounds__(256)
void scatter_add_kernel(const float* __restrict__ m,
                        const int* __restrict__ edges,
                        float* __restrict__ agg)
{
    long long gid = (long long)blockIdx.x * blockDim.x + threadIdx.x;
    const long long total = (long long)EDGES * 50;
    if (gid >= total) return;
    long long e = gid / 50;
    int c4 = (int)(gid - e * 50);
    int s = edges[e];
    int d = edges[(long long)EDGES + e];
    const float* srow = m + (size_t)s * OUTC;
    __builtin_prefetch(srow, 0, 3);          // global_prefetch_b8 (near)
    float4 v = ((const float4*)srow)[c4];
    float* orow = agg + (size_t)d * OUTC + (size_t)c4 * 4;
    atomicAdd(orow + 0, v.x);
    atomicAdd(orow + 1, v.y);
    atomicAdd(orow + 2, v.z);
    atomicAdd(orow + 3, v.w);
}

// Fused GRU cell elementwise update: h = (1-z)*n + z*h
__global__ __launch_bounds__(256)
void gru_kernel(float* __restrict__ h, const float* __restrict__ gi,
                const float* __restrict__ gh)
{
    long long gid = (long long)blockIdx.x * blockDim.x + threadIdx.x;
    const long long total = (long long)NODES * OUTC;
    if (gid >= total) return;
    long long n = gid / OUTC;
    int c = (int)(gid - n * OUTC);
    const float* gin = gi + (size_t)n * GATES;
    const float* ghn = gh + (size_t)n * GATES;
    float r  = 1.0f / (1.0f + expf(-(gin[c]       + ghn[c])));
    float z  = 1.0f / (1.0f + expf(-(gin[200 + c] + ghn[200 + c])));
    float nn = tanhf(gin[400 + c] + r * ghn[400 + c]);
    h[gid] = (1.0f - z) * nn + z * h[gid];
}

// ---------------------------------------------------------------------------
// Readout: conv1d(205->50,K=3,pad=1)+ReLU, pool(3,2), conv1d(50->20,K=1,pad=1),
// pool(2,2), then (zf@fc1)*(yf@fc2) -> sigmoid.
// ---------------------------------------------------------------------------
#define C1_TP 64
__global__ __launch_bounds__(256)
void conv1_kernel(const float* __restrict__ h, const float* __restrict__ x,
                  const float* __restrict__ w, const float* __restrict__ bsb,
                  float* __restrict__ out, int L, int mode)
{
    __shared__ float sin_[NPG][C1_TP + 2];   // 54.1 KB LDS
    const int b  = blockIdx.y;
    const int p0 = blockIdx.x * C1_TP;
    for (int i = threadIdx.x; i < NPG * (C1_TP + 2); i += 256) {
        int c = i / (C1_TP + 2);
        int j = i - c * (C1_TP + 2);
        int f = p0 - 1 + j;
        float v = 0.0f;
        if (f >= 0 && f < L) {
            size_t node = (size_t)b * NPG + c;
            if (mode == 0) v = (f < OUTC) ? h[node * OUTC + f]
                                          : x[node * EMB + (f - OUTC)];
            else           v = h[node * OUTC + f];
        }
        sin_[c][j] = v;
    }
    __syncthreads();
    for (int i = threadIdx.x; i < 50 * C1_TP; i += 256) {
        int o  = i / C1_TP;
        int pl = i - o * C1_TP;
        int p  = p0 + pl;
        if (p >= L) continue;
        float acc = bsb[o];
        const float* wo = w + (size_t)o * NPG * 3;
        for (int c = 0; c < NPG; ++c) {
            acc += wo[c*3 + 0] * sin_[c][pl]
                 + wo[c*3 + 1] * sin_[c][pl + 1]
                 + wo[c*3 + 2] * sin_[c][pl + 2];
        }
        out[((size_t)b * 50 + o) * L + p] = fmaxf(acc, 0.0f);  // fused ReLU
    }
}

__global__ __launch_bounds__(256)
void maxpool_kernel(const float* __restrict__ in, float* __restrict__ out,
                    int L, int Lo, int k, int s, long long total)
{
    long long gid = (long long)blockIdx.x * blockDim.x + threadIdx.x;
    if (gid >= total) return;
    long long rest = gid / Lo;
    int p = (int)(gid - rest * Lo);
    const float* ip = in + rest * (long long)L + (long long)p * s;
    float v = ip[0];
    for (int j = 1; j < k; ++j) v = fmaxf(v, ip[j]);
    out[gid] = v;
}

__global__ __launch_bounds__(256)
void conv2_kernel(const float* __restrict__ in, const float* __restrict__ w,
                  const float* __restrict__ bsb, float* __restrict__ out, int Lp)
{
    const int Lo = Lp + 2;
    const long long total = (long long)NGRAPH * 20 * Lo;
    long long gid = (long long)blockIdx.x * blockDim.x + threadIdx.x;
    if (gid >= total) return;
    long long rest = gid / Lo;
    int p = (int)(gid - rest * Lo);
    int o = (int)(rest % 20);
    long long b = rest / 20;
    float acc = bsb[o];
    int q = p - 1;
    if (q >= 0 && q < Lp) {
        const float* ip = in + ((size_t)b * 50) * Lp + q;
        for (int c = 0; c < 50; ++c) acc += w[o * 50 + c] * ip[(size_t)c * Lp];
    }
    out[gid] = acc;
}

__global__ __launch_bounds__(256)
void head_kernel(const float* __restrict__ zf, const float* __restrict__ yf,
                 const float* __restrict__ fc1w, const float* __restrict__ fc1b,
                 const float* __restrict__ fc2w, const float* __restrict__ fc2b,
                 float* __restrict__ out)
{
    __shared__ float red[256];
    const int b = blockIdx.x;
    float s1 = 0.0f, s2 = 0.0f;
    for (int i = threadIdx.x; i < 4840; i += 256) s1 += zf[(size_t)b*4840 + i] * fc1w[i];
    for (int i = threadIdx.x; i < 1000; i += 256) s2 += yf[(size_t)b*1000 + i] * fc2w[i];
    red[threadIdx.x] = s1;
    __syncthreads();
    for (int st = 128; st > 0; st >>= 1) {
        if (threadIdx.x < st) red[threadIdx.x] += red[threadIdx.x + st];
        __syncthreads();
    }
    float d1 = red[0];
    __syncthreads();
    red[threadIdx.x] = s2;
    __syncthreads();
    for (int st = 128; st > 0; st >>= 1) {
        if (threadIdx.x < st) red[threadIdx.x] += red[threadIdx.x + st];
        __syncthreads();
    }
    if (threadIdx.x == 0) {
        float r = (d1 + fc1b[0]) * (red[0] + fc2b[0]);
        out[b] = 1.0f / (1.0f + expf(-r));
    }
}

static inline unsigned b1d(long long n, int bs) { return (unsigned)((n + bs - 1) / bs); }

extern "C" void kernel_launch(void* const* d_in, const int* in_sizes, int n_in,
                              void* d_out, int out_size, void* d_ws, size_t ws_size,
                              hipStream_t stream)
{
    (void)in_sizes; (void)n_in; (void)out_size; (void)ws_size;

    const float* x      = (const float*)d_in[0];
    const int*   edges  = (const int*)d_in[1];     // [2, E]
    const float* enc_w  = (const float*)d_in[2];   // [768, 101]
    const float* enc_b  = (const float*)d_in[3];
    const float* ggnn_w = (const float*)d_in[4];   // [6, 200, 200]
    const float* w_ih   = (const float*)d_in[5];   // [600, 200]
    const float* w_hh   = (const float*)d_in[6];   // [600, 200]
    const float* b_ih   = (const float*)d_in[7];
    const float* b_hh   = (const float*)d_in[8];
    const float* c1w    = (const float*)d_in[9];   // [50, 205, 3]
    const float* c1b    = (const float*)d_in[10];
    const float* c2w    = (const float*)d_in[11];  // [20, 50, 1]
    const float* c2b    = (const float*)d_in[12];
    const float* fc1w   = (const float*)d_in[13];  // [4840, 1]
    const float* fc1b   = (const float*)d_in[14];
    const float* fc2w   = (const float*)d_in[15];  // [1000, 1]
    const float* fc2b   = (const float*)d_in[16];
    float* out = (float*)d_out;

    // ---- Workspace carve-up: fp32 region then f16 region (~0.96 GB total)
    float* ws  = (float*)d_ws;
    float* h   = ws;                               // N*200
    float* m   = h   + (size_t)NODES * OUTC;       // N*200
    float* agg = m   + (size_t)NODES * OUTC;       // N*200
    float* gi  = agg + (size_t)NODES * OUTC;       // N*600
    float* gh  = gi  + (size_t)NODES * GATES;      // N*600
    _Float16* xh    = (_Float16*)(gh + (size_t)NODES * GATES);  // N*768 f16
    _Float16* hh    = xh    + (size_t)NODES * EMB;              // N*224 f16
    _Float16* aggh  = hh    + (size_t)NODES * KPAD;             // N*224 f16
    _Float16* encp  = aggh  + (size_t)NODES * KPAD;             // 128*768
    _Float16* ggnnp = encp  + (size_t)128 * 768;                // 6*256*224
    _Float16* wihp  = ggnnp + (size_t)6 * 256 * KPAD;           // 640*224
    _Float16* whhp  = wihp  + (size_t)640 * KPAD;               // 640*224
    // Readout scratch aliases GGNN scratch (GGNN is done before readout):
    float* c1 = gh;                 // up to 512*50*968 = 24.78M floats
    float* p1 = m;                  // up to 512*50*483 = 12.36M
    float* c2 = agg;                // up to 512*20*485 =  4.97M
    float* zf = gi;                 // 512*4840
    float* yf = gi + 4000000;       // 512*1000

    // ---- Pack weights (f16, B^T layout, zero-padded)
    pack_bt_kernel<<<b1d((long long)128*768, 256), 256, 0, stream>>>(
        enc_w, COMP, 1, encp, 768, 128, EMB, COMP);
    for (int l = 0; l < 6; ++l) {
        pack_bt_kernel<<<b1d((long long)256*KPAD, 256), 256, 0, stream>>>(
            ggnn_w + (size_t)l * OUTC * OUTC, OUTC, 1,
            ggnnp + (size_t)l * 256 * KPAD, KPAD, 256, OUTC, OUTC);
    }
    pack_bt_kernel<<<b1d((long long)640*KPAD, 256), 256, 0, stream>>>(
        w_ih, 1, OUTC, wihp, KPAD, 640, OUTC, GATES);
    pack_bt_kernel<<<b1d((long long)640*KPAD, 256), 256, 0, stream>>>(
        w_hh, 1, OUTC, whhp, KPAD, 640, OUTC, GATES);

    // ---- x -> f16 once
    cvt_f16_kernel<<<b1d((long long)NODES*EMB/8, 256), 256, 0, stream>>>(
        x, xh, (long long)NODES * EMB / 8);

    // ---- h = 0, then encoder GEMM writes relu(x@enc_w + enc_b) to cols [0,101)
    fill_kernel<<<b1d((long long)NODES*OUTC, 256), 256, 0, stream>>>(
        h, 0.0f, (long long)NODES * OUTC);
    {
        dim3 g(NODES / GBM, (COMP + GBN - 1) / GBN);   // (820, 2)
        wmma_gemm_f16<<<g, 256, 0, stream>>>(xh, EMB, encp, EMB, enc_b,
                                             h, OUTC, EMB, COMP, 1);
    }

    // ---- 6 GGNN layers
    for (int l = 0; l < 6; ++l) {
        cvt_pad224_kernel<<<b1d((long long)NODES*(KPAD/8), 256), 256, 0, stream>>>(h, hh);
        {   // m = h @ W_l
            dim3 g(NODES / GBM, (OUTC + GBN - 1) / GBN);   // (820, 4)
            wmma_gemm_f16<<<g, 256, 0, stream>>>(hh, KPAD,
                                                 ggnnp + (size_t)l * 256 * KPAD, KPAD,
                                                 nullptr, m, OUTC, KPAD, OUTC, 0);
        }
        fill_kernel<<<b1d((long long)NODES*OUTC, 256), 256, 0, stream>>>(
            agg, 0.0f, (long long)NODES * OUTC);
        scatter_add_kernel<<<b1d((long long)EDGES*50, 256), 256, 0, stream>>>(
            m, edges, agg);
        cvt_pad224_kernel<<<b1d((long long)NODES*(KPAD/8), 256), 256, 0, stream>>>(agg, aggh);
        {   // gi = agg @ W_ih^T + b_ih ; gh = h @ W_hh^T + b_hh
            dim3 g(NODES / GBM, (GATES + GBN - 1) / GBN);  // (820, 10)
            wmma_gemm_f16<<<g, 256, 0, stream>>>(aggh, KPAD, wihp, KPAD, b_ih,
                                                 gi, GATES, KPAD, GATES, 0);
            wmma_gemm_f16<<<g, 256, 0, stream>>>(hh, KPAD, whhp, KPAD, b_hh,
                                                 gh, GATES, KPAD, GATES, 0);
        }
        gru_kernel<<<b1d((long long)NODES*OUTC, 256), 256, 0, stream>>>(h, gi, gh);
    }

    // ---- z path: concat(h,x) -> conv1+relu -> pool(3,2) -> conv2 -> pool(2,2)
    {
        dim3 g((968 + C1_TP - 1) / C1_TP, NGRAPH);
        conv1_kernel<<<g, 256, 0, stream>>>(h, x, c1w, c1b, c1, 968, 0);
    }
    maxpool_kernel<<<b1d((long long)NGRAPH*50*483, 256), 256, 0, stream>>>(
        c1, p1, 968, 483, 3, 2, (long long)NGRAPH * 50 * 483);
    conv2_kernel<<<b1d((long long)NGRAPH*20*485, 256), 256, 0, stream>>>(
        p1, c2w, c2b, c2, 483);
    maxpool_kernel<<<b1d((long long)NGRAPH*20*242, 256), 256, 0, stream>>>(
        c2, zf, 485, 242, 2, 2, (long long)NGRAPH * 20 * 242);

    // ---- y path: h -> conv1+relu -> pool(3,2) -> conv2 -> pool(2,2)
    {
        dim3 g((200 + C1_TP - 1) / C1_TP, NGRAPH);
        conv1_kernel<<<g, 256, 0, stream>>>(h, x, c1w, c1b, c1, 200, 1);
    }
    maxpool_kernel<<<b1d((long long)NGRAPH*50*99, 256), 256, 0, stream>>>(
        c1, p1, 200, 99, 3, 2, (long long)NGRAPH * 50 * 99);
    conv2_kernel<<<b1d((long long)NGRAPH*20*101, 256), 256, 0, stream>>>(
        p1, c2w, c2b, c2, 99);
    maxpool_kernel<<<b1d((long long)NGRAPH*20*50, 256), 256, 0, stream>>>(
        c2, yf, 101, 50, 2, 2, (long long)NGRAPH * 20 * 50);

    head_kernel<<<NGRAPH, 256, 0, stream>>>(zf, yf, fc1w, fc1b, fc2w, fc2b, out);
}
